// Rope_DDAK_74620761801324
// MI455X (gfx1250) — matrix operand order repaired and checked by reading, not compiled, not run on hardware
//
#include <hip/hip_runtime.h>
#include <hip/hip_bf16.h>
#include <math.h>

// ---------------- CDNA5 WMMA types ----------------
typedef __attribute__((ext_vector_type(16))) __bf16 v16bf;
typedef __attribute__((ext_vector_type(8)))  __bf16 v8bf;
typedef __attribute__((ext_vector_type(8)))  float  v8f;
typedef __attribute__((ext_vector_type(4)))  unsigned int u32x4;
typedef __attribute__((ext_vector_type(8)))  unsigned int u32x8;

static __device__ __forceinline__ v8f wmma_bf16(v16bf a, v16bf b, v8f c) {
  // D = A(16x32 bf16) * B(32x16 bf16) + C(16x16 f32)
  return __builtin_amdgcn_wmma_f32_16x16x32_bf16(false, a, false, b, (short)0, c,
                                                 false, false);
}

#define Bc 2
#define Sc 2048
#define Dc 1024
#define Hc 16
#define DHc 64

// ---------------- fp32 -> bf16 convert ----------------
__global__ __launch_bounds__(256) void cvt_bf16(const float* __restrict__ src,
                                                __bf16* __restrict__ dst, int n) {
  int i = blockIdx.x * blockDim.x + threadIdx.x;
  if (i < n) dst[i] = (__bf16)src[i];
}

// ---------------- NT GEMM: C[M,N] = A[M,K] * Bw[N,K]^T, bf16 in / f32 out ----
// One wave per 32x32 output tile (2x2 WMMA register blocking): 4 WMMAs per
// 32-wide k-step reuse 2 A-frags + 2 B-frags -> ~16 FLOP/byte from L2.
__global__ __launch_bounds__(256) void gemm_nt_bf16(const __bf16* __restrict__ A,
                                                    const __bf16* __restrict__ Bw,
                                                    float* __restrict__ C,
                                                    int M, int N, int K) {
  int wave = blockIdx.x * (blockDim.x >> 5) + (threadIdx.x >> 5);
  int lane = threadIdx.x & 31;
  int col  = lane & 15;
  int hi   = (lane >> 4) & 1;
  int tilesN = N >> 5;              // 32-wide tiles
  int tm = wave / tilesN;
  int tn = wave - tm * tilesN;

  const __bf16* a0 = A  + (size_t)(tm * 32 + col) * K;       // rows m, m+16
  const __bf16* a1 = a0 + (size_t)16 * K;
  const __bf16* b0 = Bw + (size_t)(tn * 32 + col) * K;       // cols n, n+16
  const __bf16* b1 = b0 + (size_t)16 * K;

  v8f acc00 = {}, acc01 = {}, acc10 = {}, acc11 = {};
  for (int k0 = 0; k0 < K; k0 += 32) {
    v16bf af0, af1, bf0, bf1;
    af0 = *(const v16bf*)(a0 + k0 + hi * 16);
    af1 = *(const v16bf*)(a1 + k0 + hi * 16);
    bf0 = *(const v16bf*)(b0 + k0 + hi * 16);
    bf1 = *(const v16bf*)(b1 + k0 + hi * 16);
    __builtin_prefetch(a0 + k0 + 256, 0, 1);
    __builtin_prefetch(b0 + k0 + 256, 0, 1);
    acc00 = wmma_bf16(af0, bf0, acc00);
    acc01 = wmma_bf16(af0, bf1, acc01);
    acc10 = wmma_bf16(af1, bf0, acc10);
    acc11 = wmma_bf16(af1, bf1, acc11);
  }
  float* c0 = C + (size_t)(tm * 32 + 8 * hi) * N + tn * 32 + col;
  float* c1 = c0 + (size_t)16 * N;
#pragma unroll
  for (int r = 0; r < 8; ++r) {
    c0[(size_t)r * N]      = acc00[r];
    c0[(size_t)r * N + 16] = acc01[r];
    c1[(size_t)r * N]      = acc10[r];
    c1[(size_t)r * N + 16] = acc11[r];
  }
}

// ---------------- RoPE + prep ----------------
// One wave per (b,h,s) row; lane = d in [0,32), pair (d, d+32).
__global__ __launch_bounds__(256) void rope_prep(const float* __restrict__ Qf,
                                                 const float* __restrict__ Kf,
                                                 const float* __restrict__ Vf,
                                                 __bf16* __restrict__ qbf,
                                                 __bf16* __restrict__ kbf,
                                                 float* __restrict__ kret,
                                                 __bf16* __restrict__ vt,
                                                 float* __restrict__ extra) {
  int wave = blockIdx.x * (blockDim.x >> 5) + (threadIdx.x >> 5);
  int lane = threadIdx.x & 31;
  int s  = wave & (Sc - 1);
  int bh = wave >> 11;               // b*H + h, 0..31
  int h = bh & (Hc - 1), b = bh >> 4;
  int d = lane;

  float f = __expf(-(float)d * 0.28782313662425572f);  // 10000^(-2d/64)
  float ang = (float)s * f;
  float sn, c;
  __sincosf(ang, &sn, &c);

  size_t rowx = ((size_t)b * Sc + s) * Dc + h * DHc;
  float qlo = Qf[rowx + d], qhi = Qf[rowx + 32 + d];
  float klo = Kf[rowx + d], khi = Kf[rowx + 32 + d];
  float qrlo = qlo * c - qhi * sn, qrhi = qhi * c + qlo * sn;
  float krlo = klo * c - khi * sn, krhi = khi * c + klo * sn;
  float k1lo = krlo * c - krhi * sn, k1hi = krhi * c + krlo * sn; // rope twice

  float es = qrlo * k1lo + qrhi * k1hi;
#pragma unroll
  for (int off = 1; off < 32; off <<= 1) es += __shfl_xor(es, off, 32);

  size_t rhead = ((size_t)bh * Sc + s) * DHc;
  qbf[rhead + d]      = (__bf16)qrlo;
  qbf[rhead + 32 + d] = (__bf16)qrhi;
  kbf[rhead + d]      = (__bf16)krlo;
  kbf[rhead + 32 + d] = (__bf16)krhi;
  kret[rhead + d]      = krlo;
  kret[rhead + 32 + d] = krhi;

  float vlo = Vf[rowx + d], vhi = Vf[rowx + 32 + d];
  vt[((size_t)bh * DHc + d)      * Sc + s] = (__bf16)vlo;
  vt[((size_t)bh * DHc + 32 + d) * Sc + s] = (__bf16)vhi;

  if (lane == 0) extra[(size_t)bh * Sc + s] = es * 0.125f;
}

// ---------------- Flash attention (causal + 1 diagonal extra key) ----------
// One wave per (b,h, 16-query tile). Per 32-key block: the Tensor Data Mover
// fetches the V tile (64 d x 32 keys = 4KB) into LDS asynchronously while the
// wave runs QK^T WMMAs + online softmax; after s_wait_tensorcnt the P*V WMMAs
// read B-frags straight from LDS.
__global__ __launch_bounds__(256) void attn_kernel(const __bf16* __restrict__ qbf,
                                                   const __bf16* __restrict__ kbf,
                                                   const __bf16* __restrict__ vt,
                                                   const float* __restrict__ extra,
                                                   const float* __restrict__ V1f,
                                                   float* __restrict__ out) {
  __shared__ __align__(64) __bf16 plds[8 * 16 * 32];   // P tiles, 1KB/wave
  __shared__ __align__(64) __bf16 vbuf[8 * 64 * 32];   // TDM V tiles, 4KB/wave
  int wib  = threadIdx.x >> 5;
  int wave = blockIdx.x * 8 + wib;
  int lane = threadIdx.x & 31;
  int col  = lane & 15;
  int hi   = (lane >> 4) & 1;
  int qt = wave & 127;
  int bh = wave >> 7;
  int h = bh & 15, b = bh >> 4;
  int qg0 = qt * 16;

  // Preload Q A-frags for both 32-wide d-steps (DH = 64)
  const __bf16* qrow = qbf + ((size_t)bh * Sc + qg0 + col) * DHc;
  v16bf aq0, aq1;
  aq0 = *(const v16bf*)(qrow + hi * 16);
  aq1 = *(const v16bf*)(qrow + 32 + hi * 16);

  float m[8], l[8];
  v8f o0 = {}, o1 = {}, o2 = {}, o3 = {};
#pragma unroll
  for (int r = 0; r < 8; ++r) { m[r] = -1e30f; l[r] = 0.0f; }

  __bf16* pw = &plds[wib * 512];
  __bf16* lv = &vbuf[wib * 2048];
  unsigned int ldsoff = (unsigned int)(size_t)lv;  // LDS offset = low 32 bits

  // Loop-invariant TDM descriptor words (ISA 8.3/8.4): 2D tile, data_size=2B,
  // tile 32(keys) x 64(d rows), tensor_dim0=2048, dim0_stride=2048.
  u32x4 g0;
  g0[0] = 1u;                      // count=1 valid descriptor
  g0[1] = ldsoff;                  // lds_addr
  u32x8 g1;
  g1[0] = 0x00010000u;             // data_size=1 (2 bytes), no mask/pad/iterate
  g1[1] = (2048u << 16);           // tensor_dim0[15:0] = 2048
  g1[2] = (64u << 16);             // tensor_dim1[15:0] = 64
  g1[3] = (32u << 16);             // tile_dim0 = 32
  g1[4] = 64u;                     // tile_dim1 = 64, tile_dim2 = 0
  g1[5] = 2048u;                   // tensor_dim0_stride = 2048
  g1[6] = 0u;
  g1[7] = 0u;

  const __bf16* vbase = vt + (size_t)bh * DHc * Sc;   // [d][s] for this head
  int nblk = (qg0 + 16 + 31) >> 5;

  for (int blk = 0; blk < nblk; ++blk) {
    int kb0 = blk * 32;
    // previous block's LDS reads must drain before the DMA overwrites vbuf
    asm volatile("s_wait_dscnt 0" ::: "memory");
    {
      unsigned long long ga = (unsigned long long)(size_t)(vbase + kb0);
      g0[2] = (unsigned int)ga;
      g0[3] = (unsigned int)(ga >> 32) | 0x80000000u;  // type=2 ("image")
      asm volatile("tensor_load_to_lds %0, %1" :: "s"(g0), "s"(g1) : "memory");
    }
    // ---- scores: S = Q * K^T (overlapped with the TDM transfer)
    v8f sc0 = {}, sc1 = {};
    {
      const __bf16* kp0 = kbf + ((size_t)bh * Sc + kb0 + col) * DHc;
      const __bf16* kp1 = kp0 + 16 * DHc;
      v16bf bk;
      bk = *(const v16bf*)(kp0 + hi * 16);       sc0 = wmma_bf16(aq0, bk, sc0);
      bk = *(const v16bf*)(kp0 + 32 + hi * 16);  sc0 = wmma_bf16(aq1, bk, sc0);
      bk = *(const v16bf*)(kp1 + hi * 16);       sc1 = wmma_bf16(aq0, bk, sc1);
      bk = *(const v16bf*)(kp1 + 32 + hi * 16);  sc1 = wmma_bf16(aq1, bk, sc1);
    }
    // ---- online softmax (row stats across 16-lane half groups)
#pragma unroll
    for (int r = 0; r < 8; ++r) {
      int q = qg0 + r + 8 * hi;
      float s0 = sc0[r] * 0.125f;
      float s1 = sc1[r] * 0.125f;
      if (kb0 + col > q)      s0 = -1e30f;
      if (kb0 + 16 + col > q) s1 = -1e30f;
      float mx = fmaxf(s0, s1);
#pragma unroll
      for (int off = 1; off < 16; off <<= 1) mx = fmaxf(mx, __shfl_xor(mx, off, 32));
      float mnew = fmaxf(m[r], mx);
      float fac = __expf(m[r] - mnew);
      float p0 = __expf(s0 - mnew);
      float p1 = __expf(s1 - mnew);
      float ps = p0 + p1;
#pragma unroll
      for (int off = 1; off < 16; off <<= 1) ps += __shfl_xor(ps, off, 32);
      l[r] = l[r] * fac + ps;
      m[r] = mnew;
      o0[r] *= fac; o1[r] *= fac; o2[r] *= fac; o3[r] *= fac;
      pw[(r + 8 * hi) * 32 + col]      = (__bf16)p0;
      pw[(r + 8 * hi) * 32 + 16 + col] = (__bf16)p1;
    }
    // P tile: D-layout -> A-frag layout via per-wave LDS round trip
    asm volatile("s_wait_dscnt 0" ::: "memory");
    v16bf ap;
    const __bf16* pr = pw + col * 32;
    ap = *(const v16bf*)(pr + hi * 16);
    // ---- O += P * V, V B-frags from the TDM-filled LDS tile ([d row][key])
    __builtin_amdgcn_s_wait_tensorcnt(0);
    o0 = wmma_bf16(ap, *(const v16bf*)(lv + (col)      * 32 + hi * 16), o0);
    o1 = wmma_bf16(ap, *(const v16bf*)(lv + (16 + col) * 32 + hi * 16), o1);
    o2 = wmma_bf16(ap, *(const v16bf*)(lv + (32 + col) * 32 + hi * 16), o2);
    o3 = wmma_bf16(ap, *(const v16bf*)(lv + (48 + col) * 32 + hi * 16), o3);
  }

  // ---- epilogue: diagonal extra key (k1[q], v1[q]) then normalize + store
#pragma unroll
  for (int r = 0; r < 8; ++r) {
    int q = qg0 + r + 8 * hi;
    float es = extra[(size_t)bh * Sc + q];
    float pe = __expf(es - m[r]);
    float inv = 1.0f / (l[r] + pe);
    size_t obase = ((size_t)b * Sc + q) * Dc + h * DHc + col;
    out[obase]      = (o0[r] + pe * V1f[obase])      * inv;
    out[obase + 16] = (o1[r] + pe * V1f[obase + 16]) * inv;
    out[obase + 32] = (o2[r] + pe * V1f[obase + 32]) * inv;
    out[obase + 48] = (o3[r] + pe * V1f[obase + 48]) * inv;
  }
}

// ---------------- host launcher ----------------
extern "C" void kernel_launch(void* const* d_in, const int* in_sizes, int n_in,
                              void* d_out, int out_size, void* d_ws, size_t ws_size,
                              hipStream_t stream) {
  (void)in_sizes; (void)n_in; (void)out_size; (void)ws_size;
  const float* x  = (const float*)d_in[0];
  const float* y  = (const float*)d_in[1];
  const float* wq = (const float*)d_in[2];
  /* d_in[3] = wk is UNUSED by the reference (K = y directly) */
  const float* wv = (const float*)d_in[4];
  const float* wo = (const float*)d_in[5];

  float* out  = (float*)d_out;                       // (B,S,D)
  float* kret = out + (size_t)Bc * Sc * Dc;          // (B,H,S,DH)

  const size_t NX = (size_t)Bc * Sc * Dc;            // 4,194,304
  const size_t NW = (size_t)Dc * Dc;                 // 1,048,576

  char* w = (char*)d_ws;
  size_t off = 0;
  auto take = [&](size_t bytes) { void* p = w + off; off += bytes; return p; };
  __bf16* xbf  = (__bf16*)take(2 * NX);
  __bf16* ybf  = (__bf16*)take(2 * NX);
  __bf16* wqbf = (__bf16*)take(2 * NW);
  __bf16* wvbf = (__bf16*)take(2 * NW);
  __bf16* wobf = (__bf16*)take(2 * NW);
  float*  Qf   = (float*)take(4 * NX);
  float*  Vf   = (float*)take(4 * NX);
  float*  V1f  = (float*)take(4 * NX);
  __bf16* qbf  = (__bf16*)take(2 * NX);
  __bf16* kbf  = (__bf16*)take(2 * NX);
  __bf16* vtb  = (__bf16*)take(2 * NX);
  float*  extra = (float*)take(4 * (size_t)Bc * Hc * Sc);

  // 1) fp32 -> bf16 inputs
  cvt_bf16<<<(int)((NX + 255) / 256), 256, 0, stream>>>(x, xbf, (int)NX);
  cvt_bf16<<<(int)((NX + 255) / 256), 256, 0, stream>>>(y, ybf, (int)NX);
  cvt_bf16<<<(int)((NW + 255) / 256), 256, 0, stream>>>(wq, wqbf, (int)NW);
  cvt_bf16<<<(int)((NW + 255) / 256), 256, 0, stream>>>(wv, wvbf, (int)NW);
  cvt_bf16<<<(int)((NW + 255) / 256), 256, 0, stream>>>(wo, wobf, (int)NW);

  // 2) WMMA GEMMs: Q = x*wq^T, V = y*wv^T, V1 = x*wo^T (32x32 tile per wave)
  const int M = Bc * Sc, N = Dc, K = Dc;             // 4096 x 1024 x 1024
  const int gemmBlocks = (M >> 5) * (N >> 5) / 8;    // 8 waves/block
  gemm_nt_bf16<<<gemmBlocks, 256, 0, stream>>>(xbf, wqbf, Qf,  M, N, K);
  gemm_nt_bf16<<<gemmBlocks, 256, 0, stream>>>(ybf, wvbf, Vf,  M, N, K);
  gemm_nt_bf16<<<gemmBlocks, 256, 0, stream>>>(xbf, wobf, V1f, M, N, K);

  // 3) RoPE + head split + V transpose + extra-score; also writes k_ret output
  rope_prep<<<(Bc * Hc * Sc) / 8, 256, 0, stream>>>(Qf, y, Vf, qbf, kbf, kret,
                                                    vtb, extra);

  // 4) Flash attention with TDM-fed V tiles + diagonal extra key
  attn_kernel<<<(Bc * Hc * (Sc / 16)) / 8, 256, 0, stream>>>(qbf, kbf, vtb, extra,
                                                             V1f, out);
}
